// DigitConvolutionalModel_48026324304143
// MI455X (gfx1250) — matrix operand order repaired and checked by reading, not compiled, not run on hardware
//
#include <hip/hip_runtime.h>

// Fused DigitConvolutionalModel forward for MI455X (gfx1250, wave32 + WMMA + TDM).
// Prep kernel: fc1_w/fc2_w fp32 -> f16, zero-padded, into d_ws (run every call).
// Main kernel: one block = 16 samples; TDM DMAs the x tile into LDS; 8 waves do
// fc1 with v_wmma_f32_16x16x32_f16; fc2 is a 4-tile WMMA epilogue.

typedef __attribute__((ext_vector_type(16))) _Float16     v16h;
typedef __attribute__((ext_vector_type(8)))  _Float16     v8h;
typedef __attribute__((ext_vector_type(8)))  float        v8f;
typedef __attribute__((ext_vector_type(4)))  unsigned int v4u;
typedef __attribute__((ext_vector_type(4)))  int          v4i;
typedef __attribute__((ext_vector_type(8)))  int          v8i;

#define H_STRIDE    712   // halves per conv-out row (676 -> zero pad 704, bank pad 712)
#define HACT_STRIDE 136   // halves per fc1-act row (128 padded for banks)
#define WF1_STRIDE  704   // f16 fc1 weights: K zero-padded to 704
#define WF2_ROWS    16    // f16 fc2 weights: N zero-padded to 16

// ---- weight prep: fp32 -> f16, zero-padded, laid out for direct B-fragment loads ----
__global__ __launch_bounds__(256, 1)
void prep_weights_kernel(const float* __restrict__ fc1_w,
                         const float* __restrict__ fc2_w,
                         _Float16* __restrict__ wf1,
                         _Float16* __restrict__ wf2)
{
    const int i = blockIdx.x * 256 + threadIdx.x;
    if (i < 128 * WF1_STRIDE) {
        int n = i / WF1_STRIDE;
        int k = i - n * WF1_STRIDE;
        wf1[i] = (k < 676) ? (_Float16)fc1_w[n * 676 + k] : (_Float16)0.0f;
    }
    if (i < WF2_ROWS * 128) {
        int n = i >> 7;
        int k = i & 127;
        wf2[i] = (n < 10) ? (_Float16)fc2_w[n * 128 + k] : (_Float16)0.0f;
    }
}

__global__ __launch_bounds__(256, 1)
void mnist_fused_kernel(const float* __restrict__ x,
                        const float* __restrict__ conv_w,
                        const _Float16* __restrict__ wf1,
                        const float* __restrict__ fc1_b,
                        const _Float16* __restrict__ wf2,
                        const float* __restrict__ fc2_b,
                        float* __restrict__ out)
{
    __shared__ float    xs[16 * 784];         // 50176 B, fp32 tile via TDM
    __shared__ _Float16 hh[16 * H_STRIDE];    // 22784 B
    __shared__ _Float16 ha[16 * HACT_STRIDE]; //  4352 B

    const int  tid     = threadIdx.x;
    const long sample0 = (long)blockIdx.x * 16;

    // ---- TDM: DMA the 16 x 784 fp32 x tile into LDS (one descriptor, wave 0) ----
    if (tid == 0) {
        unsigned long long ga  = (unsigned long long)(const void*)(x + sample0 * 784);
        unsigned int       lds = (unsigned int)(unsigned long long)(void*)&xs[0];
        unsigned int       td1 = gridDim.x * 16u;   // total tensor rows (OOB bound)

        // D# group 0: count=1 | lds_addr | global_addr[56:0] | type=2
        v4u g0 = { 1u,
                   lds,
                   (unsigned int)ga,
                   (unsigned int)((ga >> 32) & 0x01FFFFFFu) | (2u << 30) };
        // D# group 1: data_size=4B(code 2); tensor_dim0=784, stride0=784,
        //             tensor_dim1=td1; tile = 784 x 16
        v8i g1 = { (int)(2u << 16),
                   (int)(784u << 16),
                   (int)((td1 & 0xFFFFu) << 16),
                   (int)((td1 >> 16) | (784u << 16)),
                   16,
                   784,
                   0, 0 };
        v4i z4 = { 0, 0, 0, 0 };
#if defined(__clang_major__) && (__clang_major__ >= 23)
        v8i z8 = { 0, 0, 0, 0, 0, 0, 0, 0 };
        __builtin_amdgcn_tensor_load_to_lds(g0, g1, z4, z4, z8, 0);
#else
        __builtin_amdgcn_tensor_load_to_lds(g0, g1, z4, z4, 0);
#endif
        __builtin_amdgcn_s_wait_tensorcnt(0);
    }

    const float w00 = conv_w[0], w01 = conv_w[1], w02 = conv_w[2];
    const float w10 = conv_w[3], w11 = conv_w[4], w12 = conv_w[5];
    const float w20 = conv_w[6], w21 = conv_w[7], w22 = conv_w[8];

    __syncthreads();

    // ---- 3x3 valid conv + ReLU -> hh (f16), K zero-padded to 704 ----
    {
        const int s = tid >> 4;       // sample 0..15
        const int l = tid & 15;
        const float* xr = &xs[s * 784];
        _Float16*    hr = &hh[s * H_STRIDE];
        for (int n = l; n < 676; n += 16) {
            int r = n / 26;
            int c = n - r * 26;
            const float* p = xr + r * 28 + c;
            float acc = p[0]  * w00 + p[1]  * w01 + p[2]  * w02
                      + p[28] * w10 + p[29] * w11 + p[30] * w12
                      + p[56] * w20 + p[57] * w21 + p[58] * w22;
            hr[n] = (_Float16)fmaxf(acc, 0.0f);
        }
        for (int n = 676 + l; n < H_STRIDE; n += 16) hr[n] = (_Float16)0.0f;
    }
    __syncthreads();

    // ---- fc1: [16 x 704] @ [704 x 128] via v_wmma_f32_16x16x32_f16 ----
    const int wave = tid >> 5;     // 0..7 : N tile
    const int lane = tid & 31;
    const int M    = lane & 15;    // A/C row (sample); also B/C column select
    const int g    = lane >> 4;    // lane group
    const int n1   = wave * 16 + M;

    v8f acc = {};
    const _Float16* hA = &hh[M * H_STRIDE];
    const _Float16* wr = wf1 + n1 * WF1_STRIDE;
    for (int t = 0; t < 22; ++t) {
        // A fragment: K runs [32t+8g .. +7] and [32t+8g+16 .. +23] (ISA 16-bit A layout)
        const int kA = 32 * t + 8 * g;
        v8h a0 = *(const v8h*)(hA + kA);
        v8h a1 = *(const v8h*)(hA + kA + 16);
        v16h a = __builtin_shufflevector(a0, a1, 0,1,2,3,4,5,6,7,8,9,10,11,12,13,14,15);

        // B fragment: K runs [32t+8g .. +7] and [32t+8g+16 .. +23] (matches A)
        const int kB = 32 * t + 8 * g;
        v8h b0 = *(const v8h*)(wr + kB);
        v8h b1 = *(const v8h*)(wr + kB + 16);
        v16h b = __builtin_shufflevector(b0, b1, 0,1,2,3,4,5,6,7,8,9,10,11,12,13,14,15);

        acc = __builtin_amdgcn_wmma_f32_16x16x32_f16(false, a, false, b,
                                                     (short)0, acc, false, false);
    }

    // bias + ReLU -> fc1 activations (f16) in LDS; row = sample, col = n1
    const float b1 = fc1_b[n1];
    #pragma unroll
    for (int r = 0; r < 8; ++r) {
        float v = fmaxf(acc[r] + b1, 0.0f);
        ha[(r + 8 * g) * HACT_STRIDE + n1] = (_Float16)v;
    }
    __syncthreads();

    // ---- fc2: [16 x 128] @ [128 x 16(pad of 10)] via WMMA ----
    // All waves compute redundantly (keeps WMMA at full EXEC); wave 0 stores.
    {
        const int n2 = M;
        const _Float16* hA2 = &ha[M * HACT_STRIDE];
        const _Float16* w2  = wf2 + n2 * 128;
        v8f acc2 = {};
        #pragma unroll
        for (int t = 0; t < 4; ++t) {
            const int kA = 32 * t + 8 * g;
            v8h a0 = *(const v8h*)(hA2 + kA);
            v8h a1 = *(const v8h*)(hA2 + kA + 16);
            v16h a = __builtin_shufflevector(a0, a1, 0,1,2,3,4,5,6,7,8,9,10,11,12,13,14,15);

            const int kB = 32 * t + 8 * g;
            v8h b0 = *(const v8h*)(w2 + kB);
            v8h b1 = *(const v8h*)(w2 + kB + 16);
            v16h b = __builtin_shufflevector(b0, b1, 0,1,2,3,4,5,6,7,8,9,10,11,12,13,14,15);

            acc2 = __builtin_amdgcn_wmma_f32_16x16x32_f16(false, a, false, b,
                                                          (short)0, acc2, false, false);
        }

        if (wave == 0 && n2 < 10) {
            const float b2 = fc2_b[n2];
            #pragma unroll
            for (int r = 0; r < 8; ++r) {
                out[(sample0 + r + 8 * g) * 10 + n2] = acc2[r] + b2;
            }
        }
    }
}

extern "C" void kernel_launch(void* const* d_in, const int* in_sizes, int n_in,
                              void* d_out, int out_size, void* d_ws, size_t ws_size,
                              hipStream_t stream) {
    const float* x      = (const float*)d_in[0];
    const float* conv_w = (const float*)d_in[1];
    const float* fc1_w  = (const float*)d_in[2];
    const float* fc1_b  = (const float*)d_in[3];
    const float* fc2_w  = (const float*)d_in[4];
    const float* fc2_b  = (const float*)d_in[5];
    float* out = (float*)d_out;

    // Workspace layout: wf1 = f16[128][704], then wf2 = f16[16][128]  (~184 KB)
    _Float16* wf1 = (_Float16*)d_ws;
    _Float16* wf2 = wf1 + 128 * WF1_STRIDE;

    const int prepN = 128 * WF1_STRIDE;
    prep_weights_kernel<<<(prepN + 255) / 256, 256, 0, stream>>>(fc1_w, fc2_w, wf1, wf2);

    const int B = in_sizes[0] / 784;        // 65536
    mnist_fused_kernel<<<dim3(B / 16), dim3(256), 0, stream>>>(
        x, conv_w, wf1, fc1_b, wf2, fc2_b, out);
}
